// UncoditionalDecoder_49563922596234
// MI455X (gfx1250) — matrix operand order repaired and checked by reading, not compiled, not run on hardware
//
#include <hip/hip_runtime.h>
#include <math.h>

// Tacotron-style decoder for MI455X (gfx1250, wave32, WMMA).
// Workspace requirement: ~40 MB (bf16 weights ~15 MB + activations ~22 MB + state).
//
// d_in order (reference setup_inputs insertion order):
//  0 lengths(i32,16) 1 mels(f32,16*200*80)
//  2..4 prenet W (512x80, 512x512, 512x512)
//  5..7 encoder W (512x80, 512x512, 512x512)
//  8 ar.Wih(2048x1024) 9 ar.Whh(2048x512) 10 ar.b(2048)
// 11 dr.Wih(2048x1024) 12 dr.Whh(2048x512) 13 dr.b(2048)
// 14 Wq(128x512) 15 Wm(128x512) 16 conv(32x2x31) 17 Wl(128x32) 18 v(128)
// 19 out.W(80x1024) 20 out.b(80)
// d_out: mel_seq (16,200,80) f32  followed by  align (16,200,200) f32.

#define B_    16
#define T_    200
#define NMEL_ 80
#define NMELP 96
#define HID_  512
#define ATT_  128
#define NF_   32
#define KC_   31
#define ZO_   0.1f

typedef __attribute__((ext_vector_type(16))) __bf16        v16bf;
typedef __attribute__((ext_vector_type(8)))  float         v8f;
typedef __attribute__((ext_vector_type(4)))  unsigned int  v4u;

union FragBF { v16bf v; v4u q[2]; };

__device__ inline unsigned short f2bf(float f) {
  unsigned u = __float_as_uint(f);
  u = u + 0x7fffu + ((u >> 16) & 1u);          // round-to-nearest-even
  return (unsigned short)(u >> 16);
}

// One K=32 WMMA step. pa/pb are row base pointers (no lane-half offset).
// A frag (16x32 bf16): lane m, halves = A[m][8h+0..7] , A[m][16+8h+0..7]
// B frag (32x16 bf16): lane n, halves = W[n][8h+0..7] , W[n][16+8h+0..7]  (W row-major NxK, B = W^T)
__device__ inline v8f wmma_step32(v8f acc, const unsigned short* pa,
                                  const unsigned short* pb, int half) {
  FragBF a, b;
  a.q[0] = *(const v4u*)(pa + 8 * half);
  a.q[1] = *(const v4u*)(pa + 8 * half + 16);
  b.q[0] = *(const v4u*)(pb + 8 * half);
  b.q[1] = *(const v4u*)(pb + 8 * half + 16);
  return __builtin_amdgcn_wmma_f32_16x16x32_bf16(false, a.v, false, b.v,
                                                 (short)0, acc, false, false);
}

__device__ inline v8f wmma_kloop(v8f acc, const unsigned short* pa,
                                 const unsigned short* pb, int klen, int half) {
  for (int kk = 0; kk < klen; kk += 32)
    acc = wmma_step32(acc, pa + kk, pb + kk, half);
  return acc;
}

// ---------------------------------------------------------------- utilities
__global__ void convert_pad(const float* __restrict__ src, unsigned short* __restrict__ dst,
                            int N, int Ks, int Kd) {
  int i = blockIdx.x * blockDim.x + threadIdx.x;
  if (i >= N * Kd) return;
  int r = i / Kd, c = i % Kd;
  float v = (c < Ks) ? src[(size_t)r * Ks + c] : 0.0f;
  dst[i] = f2bf(v);
}

__global__ void zero_u32(unsigned int* p, int n) {
  int i = blockIdx.x * blockDim.x + threadIdx.x;
  if (i < n) p[i] = 0u;
}

__device__ inline float rand_uniform(unsigned idx) {
  unsigned x = idx * 747796405u + 2891336453u;
  x = ((x >> ((x >> 28) + 4u)) ^ x) * 277803737u;
  x = (x >> 22) ^ x;
  return (float)(x >> 8) * (1.0f / 16777216.0f);
}

// dec_in (B,T,NMELP) bf16: t==0 -> pseudo-random frame0, else mels[b][t-1]
__global__ void build_decin(const float* __restrict__ mels, unsigned short* __restrict__ out) {
  int i = blockIdx.x * blockDim.x + threadIdx.x;
  if (i >= B_ * T_ * NMELP) return;
  int c = i % NMELP, bt = i / NMELP;
  int t = bt % T_, b = bt / T_;
  float v = 0.0f;
  if (c < NMEL_)
    v = (t == 0) ? rand_uniform(b * NMEL_ + c)
                 : mels[((size_t)b * T_ + (t - 1)) * NMEL_ + c];
  out[i] = f2bf(v);
}

// ------------------------------------------------------- generic bf16 GEMM
// C(MxN) = act( A(MxK,bf16) * W(NxK,bf16)^T ); 8 waves/WG, one 16x16 tile/wave.
__global__ __launch_bounds__(256) void gemm_bf16(
    const unsigned short* __restrict__ A, int lda,
    const unsigned short* __restrict__ W, int ldw,
    float* __restrict__ Cf, unsigned short* __restrict__ Cb, int ldc,
    int M, int N, int K, int relu) {
  int tiles = (M >> 4) * (N >> 4);
  int tile = blockIdx.x * 8 + (threadIdx.x >> 5);
  if (tile >= tiles) return;
  int ntn = N >> 4;
  int tm = tile / ntn, tn = tile % ntn;
  int lane = threadIdx.x & 31;
  int n = lane & 15, half = lane >> 4;
  v8f acc;
#pragma unroll
  for (int r = 0; r < 8; r++) acc[r] = 0.0f;
  const unsigned short* pa = A + (size_t)(tm * 16 + n) * lda;
  const unsigned short* pb = W + (size_t)(tn * 16 + n) * ldw;
  acc = wmma_kloop(acc, pa, pb, K, half);
#pragma unroll
  for (int r = 0; r < 8; r++) {
    int row = tm * 16 + 8 * half + r;
    int col = tn * 16 + n;
    float v = acc[r];
    if (relu) v = fmaxf(v, 0.0f);
    if (Cf) Cf[(size_t)row * ldc + col] = v;
    if (Cb) Cb[(size_t)row * ldc + col] = f2bf(v);
  }
}

// ------------------------------------------------- fused LSTM step (M=16=B)
// z = [A1|A2] @ Wih^T + H @ Whh^T + b ; gates -> c2,h2 ; zoneout vs hOld/cOld.
// grid 32 WGs (16 hidden units each), 4 waves = 4 gates, one WMMA tile each.
__global__ __launch_bounds__(128) void lstm_step(
    const unsigned short* __restrict__ A1, int lda1, int K1,
    const unsigned short* __restrict__ A2, int K2,
    const unsigned short* __restrict__ Wih,
    const unsigned short* __restrict__ Hbf,
    const unsigned short* __restrict__ Whh,
    const float* __restrict__ bias,
    const float* __restrict__ hOld, const float* __restrict__ cOld,
    float* __restrict__ hNew, float* __restrict__ cNew,
    unsigned short* __restrict__ hNewBf) {
  __shared__ float zbuf[4][16][16];              // [gate][batch][unit]
  const int j = blockIdx.x;                      // hidden-unit chunk
  const int g = threadIdx.x >> 5;                // gate = wave id
  const int lane = threadIdx.x & 31;
  const int n = lane & 15, half = lane >> 4;
  const int wrow = g * HID_ + j * 16 + n;        // weight row (output column)
  const int ldw = K1 + K2;
  float bv = bias[wrow];
  v8f acc;
#pragma unroll
  for (int r = 0; r < 8; r++) acc[r] = bv;
  acc = wmma_kloop(acc, A1 + (size_t)n * lda1, Wih + (size_t)wrow * ldw, K1, half);
  acc = wmma_kloop(acc, A2 + (size_t)n * K2, Wih + (size_t)wrow * ldw + K1, K2, half);
  acc = wmma_kloop(acc, Hbf + (size_t)n * HID_, Whh + (size_t)wrow * HID_, HID_, half);
#pragma unroll
  for (int r = 0; r < 8; r++) zbuf[g][8 * half + r][n] = acc[r];
  __syncthreads();
  for (int idx = threadIdx.x; idx < 256; idx += 128) {
    int bb = idx >> 4, u = idx & 15;
    int unit = j * 16 + u;
    float iv = zbuf[0][bb][u], fv = zbuf[1][bb][u];
    float gv = zbuf[2][bb][u], ov = zbuf[3][bb][u];
    float hP = hOld[bb * HID_ + unit], cP = cOld[bb * HID_ + unit];
    float si = 1.0f / (1.0f + __expf(-iv));
    float sf = 1.0f / (1.0f + __expf(-fv));
    float so = 1.0f / (1.0f + __expf(-ov));
    float c2 = sf * cP + si * tanhf(gv);
    float h2 = so * tanhf(c2);
    float hz = ZO_ * hP + (1.0f - ZO_) * h2;
    float cz = ZO_ * cP + (1.0f - ZO_) * c2;
    hNew[bb * HID_ + unit] = hz;
    cNew[bb * HID_ + unit] = cz;
    hNewBf[bb * HID_ + unit] = f2bf(hz);
  }
}

// ---------------------------------------- location-sensitive attention step
// One workgroup per batch row. q, conv-31 loc features, tanh energies,
// masked softmax, cumulative-weight update, context vector.
__global__ __launch_bounds__(256) void attention_step(
    int step,
    const float* __restrict__ ah2,   // 16x512
    const float* __restrict__ enc,   // (B*T)x512
    const float* __restrict__ pmem,  // (B*T)x128
    float* __restrict__ aw, float* __restrict__ aws,  // 16x200 in/out
    const int* __restrict__ lengths,
    const float* __restrict__ convw,  // 32*2*31
    const float* __restrict__ Wl,     // 128*32
    const float* __restrict__ vvec,   // 128
    const float* __restrict__ Wq,     // 128*512
    float* __restrict__ ctx, unsigned short* __restrict__ ctxbf,  // 16x512
    float* __restrict__ align_out) {  // (B,T,T)
  const int b = blockIdx.x, tid = threadIdx.x;
  __shared__ float qL[ATT_];
  __shared__ float awp[T_ + 2 * 15], awsp[T_ + 2 * 15];  // halo-padded
  __shared__ float wL[T_ + 8];
  __shared__ float red[256];
  // load cumulative weights with zero halo
  for (int i = tid; i < T_ + 30; i += 256) {
    int t = i - 15;
    float a = 0.0f, s = 0.0f;
    if (t >= 0 && t < T_) { a = aw[b * T_ + t]; s = aws[b * T_ + t]; }
    awp[i] = a; awsp[i] = s;
  }
  // q = ah2[b] @ Wq^T
  for (int a = tid; a < ATT_; a += 256) {
    float s = 0.0f;
    const float* hr = ah2 + b * HID_;
    const float* wr = Wq + (size_t)a * HID_;
    for (int k = 0; k < HID_; k++) s += hr[k] * wr[k];
    qL[a] = s;
  }
  __syncthreads();
  // energies
  const int t = tid;
  const int len = lengths[b];
  bool valid = false;
  float myE = -1e9f;
  if (t < T_) {
    float locf[NF_];
#pragma unroll 4
    for (int f = 0; f < NF_; f++) {
      const float* c0 = convw + f * (2 * KC_);
      const float* c1 = c0 + KC_;
      float s = 0.0f;
      for (int k = 0; k < KC_; k++) s += c0[k] * awp[t + k] + c1[k] * awsp[t + k];
      locf[f] = s;
    }
    float e = 0.0f;
    const float* pm = pmem + ((size_t)b * T_ + t) * ATT_;
    for (int a = 0; a < ATT_; a++) {
      float pl = 0.0f;
      const float* wl = Wl + a * NF_;
      for (int f = 0; f < NF_; f++) pl += wl[f] * locf[f];
      e += vvec[a] * tanhf(qL[a] + pl + pm[a]);
    }
    valid = (t < len) && (t <= step);
    myE = valid ? e : -1e9f;
  }
  // softmax: max then sum
  red[tid] = (t < T_) ? myE : -3.0e38f;
  __syncthreads();
  for (int s = 128; s > 0; s >>= 1) {
    if (tid < s) red[tid] = fmaxf(red[tid], red[tid + s]);
    __syncthreads();
  }
  float mx = red[0];
  __syncthreads();
  float ex = (t < T_) ? __expf(myE - mx) : 0.0f;
  red[tid] = ex;
  __syncthreads();
  for (int s = 128; s > 0; s >>= 1) {
    if (tid < s) red[tid] += red[tid + s];
    __syncthreads();
  }
  float sm = red[0];
  if (t < T_) {
    float w = valid ? (ex / sm) : 0.0f;
    wL[t] = w;
    float aw2 = awp[15 + t] + w;
    float aws2 = awsp[15 + t] + aw2;
    aw[b * T_ + t] = aw2;
    aws[b * T_ + t] = aws2;
    align_out[((size_t)b * T_ + step) * T_ + t] = aw2;
  }
  __syncthreads();
  // context = w . enc_out
  for (int c = tid; c < HID_; c += 256) {
    float s = 0.0f;
    for (int tt = 0; tt < T_; tt++) s += wL[tt] * enc[((size_t)b * T_ + tt) * HID_ + c];
    ctx[b * HID_ + c] = s;
    ctxbf[b * HID_ + c] = f2bf(s);
  }
}

// -------------------------------------------------- output projection (WMMA)
// mel_t = [dh2|ctx2] @ Wout^T + b ; 5 tiles of N (80 cols), one wave per WG.
__global__ __launch_bounds__(32) void mel_out_step(
    int step,
    const unsigned short* __restrict__ Dh, const unsigned short* __restrict__ Ctx,
    const unsigned short* __restrict__ Wo, const float* __restrict__ bias,
    float* __restrict__ mel) {
  const int tn = blockIdx.x;
  const int lane = threadIdx.x & 31;
  const int n = lane & 15, half = lane >> 4;
  const int col = tn * 16 + n;  // < 80
  v8f acc;
  float bv = bias[col];
#pragma unroll
  for (int r = 0; r < 8; r++) acc[r] = bv;
  const unsigned short* pb = Wo + (size_t)col * 1024;
  acc = wmma_kloop(acc, Dh + (size_t)n * HID_, pb, HID_, half);
  acc = wmma_kloop(acc, Ctx + (size_t)n * HID_, pb + HID_, HID_, half);
#pragma unroll
  for (int r = 0; r < 8; r++) {
    int bb = 8 * half + r;
    mel[((size_t)bb * T_ + step) * NMEL_ + col] = acc[r];
  }
}

// =========================================================== host launcher
extern "C" void kernel_launch(void* const* d_in, const int* in_sizes, int n_in,
                              void* d_out, int out_size, void* d_ws, size_t ws_size,
                              hipStream_t stream) {
  (void)in_sizes; (void)n_in; (void)out_size; (void)ws_size;
  const int*   lengths = (const int*)d_in[0];
  const float* mels    = (const float*)d_in[1];
  const float* pre0 = (const float*)d_in[2];
  const float* pre1 = (const float*)d_in[3];
  const float* pre2 = (const float*)d_in[4];
  const float* enc0 = (const float*)d_in[5];
  const float* enc1 = (const float*)d_in[6];
  const float* enc2 = (const float*)d_in[7];
  const float* arWih = (const float*)d_in[8];
  const float* arWhh = (const float*)d_in[9];
  const float* arB   = (const float*)d_in[10];
  const float* drWih = (const float*)d_in[11];
  const float* drWhh = (const float*)d_in[12];
  const float* drB   = (const float*)d_in[13];
  const float* Wq    = (const float*)d_in[14];
  const float* Wm    = (const float*)d_in[15];
  const float* convw = (const float*)d_in[16];
  const float* Wl    = (const float*)d_in[17];
  const float* vvec  = (const float*)d_in[18];
  const float* Wout  = (const float*)d_in[19];
  const float* outB  = (const float*)d_in[20];

  char* ws = (char*)d_ws;
  size_t off = 0;
  auto ALC = [&](size_t bytes) { size_t o = off; off += (bytes + 255) & ~(size_t)255; return o; };
  auto U16 = [&](size_t o) { return (unsigned short*)(ws + o); };
  auto F32 = [&](size_t o) { return (float*)(ws + o); };

  // bf16 weights (K padded to mult of 32)
  size_t oPre0 = ALC(512 * NMELP * 2), oPre1 = ALC(512 * 512 * 2), oPre2 = ALC(512 * 512 * 2);
  size_t oEnc0 = ALC(512 * NMELP * 2), oEnc1 = ALC(512 * 512 * 2), oEnc2 = ALC(512 * 512 * 2);
  size_t oArIh = ALC((size_t)2048 * 1024 * 2), oArHh = ALC((size_t)2048 * 512 * 2);
  size_t oDrIh = ALC((size_t)2048 * 1024 * 2), oDrHh = ALC((size_t)2048 * 512 * 2);
  size_t oWm   = ALC(128 * 512 * 2),           oWo   = ALC(80 * 1024 * 2);
  // activations
  const size_t R = (size_t)B_ * T_;  // 3200 rows
  size_t oDecin = ALC(R * NMELP * 2);
  size_t oH1    = ALC(R * HID_ * 2);
  size_t oH2    = ALC(R * HID_ * 2);
  size_t oPreB  = ALC(R * HID_ * 2);
  size_t oEncF  = ALC(R * HID_ * 4);
  size_t oEncB  = ALC(R * HID_ * 2);
  size_t oPmem  = ALC(R * ATT_ * 4);
  // recurrent state (zeroed each call); [2] = ping-pong parity
  size_t stateBeg = off;
  size_t oAhF = ALC(2 * B_ * HID_ * 4), oAcF = ALC(2 * B_ * HID_ * 4), oAhB = ALC(2 * B_ * HID_ * 2);
  size_t oDhF = ALC(2 * B_ * HID_ * 4), oDcF = ALC(2 * B_ * HID_ * 4), oDhB = ALC(2 * B_ * HID_ * 2);
  size_t oCtxF = ALC(B_ * HID_ * 4), oCtxB = ALC(B_ * HID_ * 2);
  size_t oAw = ALC(B_ * T_ * 4), oAws = ALC(B_ * T_ * 4);
  size_t stateBytes = off - stateBeg;

  // ---- weight conversion (fp32 -> bf16, zero K-padding)
  auto cvt = [&](const float* s, size_t d, int N, int Ks, int Kd) {
    int tot = N * Kd;
    convert_pad<<<(tot + 255) / 256, 256, 0, stream>>>(s, U16(d), N, Ks, Kd);
  };
  cvt(pre0, oPre0, 512, NMEL_, NMELP); cvt(pre1, oPre1, 512, 512, 512); cvt(pre2, oPre2, 512, 512, 512);
  cvt(enc0, oEnc0, 512, NMEL_, NMELP); cvt(enc1, oEnc1, 512, 512, 512); cvt(enc2, oEnc2, 512, 512, 512);
  cvt(arWih, oArIh, 2048, 1024, 1024); cvt(arWhh, oArHh, 2048, 512, 512);
  cvt(drWih, oDrIh, 2048, 1024, 1024); cvt(drWhh, oDrHh, 2048, 512, 512);
  cvt(Wm, oWm, 128, 512, 512);         cvt(Wout, oWo, 80, 1024, 1024);

  // ---- dec_in, state init
  {
    int tot = B_ * T_ * NMELP;
    build_decin<<<(tot + 255) / 256, 256, 0, stream>>>(mels, U16(oDecin));
    int n32 = (int)(stateBytes / 4);
    zero_u32<<<(n32 + 255) / 256, 256, 0, stream>>>((unsigned int*)(ws + stateBeg), n32);
  }

  // ---- parallel front end (WMMA GEMMs)
  auto gemm = [&](size_t a, int lda, size_t w, int ldw, float* cf, unsigned short* cb,
                  int ldc, int M, int N, int K, int relu) {
    int tiles = (M >> 4) * (N >> 4);
    gemm_bf16<<<(tiles + 7) / 8, 256, 0, stream>>>(U16(a), lda, U16(w), ldw, cf, cb, ldc,
                                                   M, N, K, relu);
  };
  // prenet: decin -> h1 -> h2 -> preB (relu, bf16 out)
  gemm(oDecin, NMELP, oPre0, NMELP, nullptr, U16(oH1), HID_, (int)R, HID_, NMELP, 1);
  gemm(oH1, HID_, oPre1, HID_, nullptr, U16(oH2), HID_, (int)R, HID_, HID_, 1);
  gemm(oH2, HID_, oPre2, HID_, nullptr, U16(oPreB), HID_, (int)R, HID_, HID_, 1);
  // encoder: decin -> h1 -> h2 -> enc (f32 + bf16)
  gemm(oDecin, NMELP, oEnc0, NMELP, nullptr, U16(oH1), HID_, (int)R, HID_, NMELP, 1);
  gemm(oH1, HID_, oEnc1, HID_, nullptr, U16(oH2), HID_, (int)R, HID_, HID_, 1);
  gemm(oH2, HID_, oEnc2, HID_, F32(oEncF), U16(oEncB), HID_, (int)R, HID_, HID_, 1);
  // pmem = enc @ Wm^T (f32)
  gemm(oEncB, HID_, oWm, HID_, F32(oPmem), nullptr, ATT_, (int)R, ATT_, HID_, 0);

  // ---- sequential scan, 4 launches per step
  float* melOut = (float*)d_out;
  float* alignOut = melOut + (size_t)B_ * T_ * NMEL_;
  const size_t hf = (size_t)B_ * HID_;  // parity stride (elements)
  for (int t = 0; t < T_; t++) {
    int pi = t & 1, ni = (t + 1) & 1;
    // attention LSTM: A = [prenet(x_t) | ctx]
    lstm_step<<<32, 128, 0, stream>>>(
        U16(oPreB) + (size_t)t * HID_, T_ * HID_, HID_,
        U16(oCtxB), HID_,
        U16(oArIh), U16(oAhB) + pi * hf, U16(oArHh), arB,
        F32(oAhF) + pi * hf, F32(oAcF) + pi * hf,
        F32(oAhF) + ni * hf, F32(oAcF) + ni * hf, U16(oAhB) + ni * hf);
    // location-sensitive attention -> ctx, aw, aws, alignments
    attention_step<<<B_, 256, 0, stream>>>(
        t, F32(oAhF) + ni * hf, F32(oEncF), F32(oPmem),
        F32(oAw), F32(oAws), lengths, convw, Wl, vvec, Wq,
        F32(oCtxF), U16(oCtxB), alignOut);
    // decoder LSTM: A = [ah2 | ctx2]
    lstm_step<<<32, 128, 0, stream>>>(
        U16(oAhB) + ni * hf, HID_, HID_,
        U16(oCtxB), HID_,
        U16(oDrIh), U16(oDhB) + pi * hf, U16(oDrHh), drB,
        F32(oDhF) + pi * hf, F32(oDcF) + pi * hf,
        F32(oDhF) + ni * hf, F32(oDcF) + ni * hf, U16(oDhB) + ni * hf);
    // mel frame
    mel_out_step<<<5, 32, 0, stream>>>(t, U16(oDhB) + ni * hf, U16(oCtxB),
                                       U16(oWo), outB, melOut);
  }
}
